// NeuralSDE_forecasting_55301998903513
// MI455X (gfx1250) — matrix operand order repaired and checked by reading, not compiled, not run on hardware
//
#include <hip/hip_runtime.h>
#include <math.h>

// ---- Problem constants (from reference) ----
#define Bdim  256
#define Tdim  256
#define Cdim  32
#define Hdim  512
#define Odim  32
#define OUTT  32

// ---- WMMA fragment types (CDNA5 / gfx1250, wave32) ----
typedef __attribute__((ext_vector_type(16))) __bf16 v16bf;
typedef __attribute__((ext_vector_type(8)))  float  v8f;
typedef __attribute__((ext_vector_type(4)))  int    v4i;

// ---- Feature detection: async global->LDS (gfx1250 ASYNCcnt path) ----
#ifndef HAVE_ASYNC
#if defined(__has_builtin)
#if __has_builtin(__builtin_amdgcn_global_load_async_to_lds_b128)
#define HAVE_ASYNC 1
#else
#define HAVE_ASYNC 0
#endif
#else
#define HAVE_ASYNC 0
#endif
#endif

// ---- Packed-weight layout (bf16 B-fragments, 32x16 tiles, 512 halves/tile) ----
// tile (kc, nt) at linear index kc*NT + nt; within a tile, lane l owns 16
// contiguous halves at l*16: lane<16 -> K=0..15, lane>=16 -> K=16..31, N=l%16.
static constexpr int OFF_WINIT = 0;                       // 32x512   -> 32 tiles
static constexpr int OFF_W1    = OFF_WINIT + 32 * 512;    // 544x512  -> 17*32 tiles
static constexpr int OFF_W2    = OFF_W1    + 544 * 512;   // 512x512
static constexpr int OFF_WG    = OFF_W2    + 512 * 512;
static constexpr int OFF_WH1   = OFF_WG    + 512 * 512;
static constexpr int OFF_WH2   = OFF_WH1   + 512 * 512;   // 512x32 -> 16*2 tiles
static constexpr int PACK_TOTAL = OFF_WH2  + 512 * 32;    // 1,097,728 halves

static constexpr int APITCH  = 560;  // halves: [y(512) | x(32)] + pad (row = 1120B)
static constexpr int HPITCH  = 528;  // halves: h / relu staging (row = 1056B)
static constexpr int DWPITCH = 520;  // floats: dW staging row pitch (2080B, 16B-aligned)
static constexpr int NTILES  = Hdim / 16;  // 32
static constexpr int TPW     = 4;          // N-tiles per wave (8 waves * 4 = 32)

// Native f32 -> bf16 (backend selects v_cvt_*bf16* on gfx1250, RTNE)
__device__ __forceinline__ __bf16 to_bf16(float f) { return (__bf16)f; }

__device__ __forceinline__ float fast_tanh(float x) {
  float e = __expf(2.0f * x);                       // v_exp_f32 (trans)
  return 1.0f - 2.0f * __builtin_amdgcn_rcpf(e + 1.0f);
}
__device__ __forceinline__ float fast_sigmoid(float x) {
  return __builtin_amdgcn_rcpf(1.0f + __expf(-x));
}

union V16U { v16bf v; uint4 q[2]; };

__device__ __forceinline__ v16bf load_frag(const __bf16* p0, const __bf16* p1) {
  V16U r;
  r.q[0] = *(const uint4*)p0;
  r.q[1] = *(const uint4*)p1;
  return r.v;
}

// A-fragment (16x32 bf16) from row-major bf16 LDS buffer.
// lane l: row = l%16; halves 0..15 = cols colBase + (l/16)*16 .. +15.
__device__ __forceinline__ v16bf loadA(const __bf16* s, int pitch, int lane, int colBase) {
  const __bf16* base = s + (lane & 15) * pitch + colBase + ((lane >> 4) << 4);
  return load_frag(base, base + 8);
}

// B-fragment (32x16 bf16) from pre-packed weights: lane's 16 halves contiguous.
__device__ __forceinline__ v16bf loadB(const __bf16* pw, int tileIdx, int lane) {
  const __bf16* p = pw + ((size_t)tileIdx << 9) + (lane << 4);
  return load_frag(p, p + 8);
}

__device__ __forceinline__ v8f wmma_bf16(v16bf a, v16bf b, v8f c) {
  return __builtin_amdgcn_wmma_f32_16x16x32_bf16(false, a, false, b, (short)0, c,
                                                 false, false);
}

#if HAVE_ASYNC
__device__ __forceinline__ void async_b128(const void* g, void* l) {
  // builtin signature: (v4i addrspace(1)*, v4i addrspace(3)*, imm offset, imm cpol)
  __attribute__((address_space(1))) void* gp =
      (__attribute__((address_space(1))) void*)g;
  __attribute__((address_space(3))) void* lp =
      (__attribute__((address_space(3))) void*)l;
  __builtin_amdgcn_global_load_async_to_lds_b128(
      (__attribute__((address_space(1))) v4i*)gp,
      (__attribute__((address_space(3))) v4i*)lp, 0, 0);
}
__device__ __forceinline__ void wait_async_le8() {
#if __has_builtin(__builtin_amdgcn_s_wait_asynccnt)
  __builtin_amdgcn_s_wait_asynccnt(8);
#else
  asm volatile("s_wait_asynccnt 0x8" ::: "memory");
#endif
}
__device__ __forceinline__ void wait_async_le0() {
#if __has_builtin(__builtin_amdgcn_s_wait_asynccnt)
  __builtin_amdgcn_s_wait_asynccnt(0);
#else
  asm volatile("s_wait_asynccnt 0x0" ::: "memory");
#endif
}
#endif

// ---------------- weight pack + dt kernel ----------------
__global__ void __launch_bounds__(256)
sde_pack(const float* __restrict__ times,
         const float* __restrict__ Wi,  const float* __restrict__ W1,
         const float* __restrict__ W2,  const float* __restrict__ Wg,
         const float* __restrict__ Wh1, const float* __restrict__ Wh2,
         float* __restrict__ hdr, __bf16* __restrict__ pw) {
  int gid = blockIdx.x * blockDim.x + threadIdx.x;
  if (gid == 0) {
    float mn = 1e30f;
    for (int i = 0; i < Tdim - 1; ++i) mn = fminf(mn, times[i + 1] - times[i]);
    float dt = fmaxf(mn, 0.001f);
    hdr[0] = dt;
    hdr[1] = sqrtf(dt);
  }
  for (int h = gid; h < PACK_TOTAL; h += gridDim.x * blockDim.x) {
    const float* src; int N, local;
    if      (h < OFF_W1)  { src = Wi;  N = 512; local = h - OFF_WINIT; }
    else if (h < OFF_W2)  { src = W1;  N = 512; local = h - OFF_W1;  }
    else if (h < OFF_WG)  { src = W2;  N = 512; local = h - OFF_W2;  }
    else if (h < OFF_WH1) { src = Wg;  N = 512; local = h - OFF_WG;  }
    else if (h < OFF_WH2) { src = Wh1; N = 512; local = h - OFF_WH1; }
    else                  { src = Wh2; N = 32;  local = h - OFF_WH2; }
    int tile   = local >> 9;
    int within = local & 511;
    int lane   = within >> 4;
    int hh     = within & 15;
    int NT     = N >> 4;
    int nt     = tile % NT;
    int kc     = tile / NT;
    int n      = nt * 16 + (lane & 15);
    int k      = kc * 32 + ((lane >> 4) << 4) + hh;
    pw[h] = to_bf16(src[(size_t)k * N + n]);
  }
}

// ---------------- persistent SDE scan + fused head ----------------
__global__ void __launch_bounds__(256)
sde_main(const float* __restrict__ coeffs, const float* __restrict__ dW,
         const float* __restrict__ b_init, const float* __restrict__ b1,
         const float* __restrict__ b2,     const float* __restrict__ bg,
         const float* __restrict__ bh1,    const float* __restrict__ bh2,
         const float* __restrict__ hdr,    const __bf16* __restrict__ pw,
         float* __restrict__ out) {
  __shared__ __bf16 sA[16 * APITCH];   // [ y (512) | x (32) ] bf16, row-major
  __shared__ __bf16 sH[16 * HPITCH];   // h / relu staging, bf16, row-major
#if HAVE_ASYNC
  __shared__ float sDW[2][16 * DWPITCH];  // double-buffered dW[k] (async-filled)
#endif

  const int tid   = threadIdx.x;
  const int lane  = tid & 31;
  const int wave  = tid >> 5;
  const int ln16  = lane & 15;
  const int hi8   = (lane >> 4) << 3;          // C-frag row offset: 0 or 8
  const int bbase = blockIdx.x * 16;           // 16 batch rows per workgroup
  const int nt0   = wave * TPW;

  const float dt   = hdr[0];
  const float sqdt = hdr[1];

  const __bf16* P_Wi  = pw + OFF_WINIT;
  const __bf16* P_W1  = pw + OFF_W1;
  const __bf16* P_W2  = pw + OFF_W2;
  const __bf16* P_Wg  = pw + OFF_WG;
  const __bf16* P_Wh1 = pw + OFF_WH1;
  const __bf16* P_Wh2 = pw + OFF_WH2;

#if HAVE_ASYNC
  // kick off dW[0] -> sDW[0] (8 x b128 per thread = 32KB block)
  {
    const float* gb = dW + (size_t)bbase * Hdim;
    float* lb = &sDW[0][0];
#pragma unroll
    for (int i = 0; i < 8; ++i) {
      int e = tid + i * 256;            // 16-byte chunk index, 128 per row
      int r = e >> 7, c4 = (e & 127) << 2;
      async_b128(gb + (size_t)r * Hdim + c4, lb + r * DWPITCH + c4);
    }
  }
#endif

  // ---- stage x0 into sA x-region ----
  for (int i = tid; i < 16 * Cdim; i += 256) {
    int r = i >> 5, c = i & 31;
    sA[r * APITCH + 512 + c] =
        to_bf16(coeffs[(size_t)(bbase + r) * Tdim * Cdim + c]);
  }
  __syncthreads();

  // ---- z0 = x0 @ W_init + b_init (K = 32, one chunk) ----
  v8f y[TPW];
  {
    v16bf a = loadA(sA, APITCH, lane, 512);
#pragma unroll
    for (int t = 0; t < TPW; ++t) {
      int nt = nt0 + t;
      v8f c; float bv = b_init[nt * 16 + ln16];
#pragma unroll
      for (int i = 0; i < 8; ++i) c[i] = bv;
      y[t] = wmma_bf16(a, loadB(P_Wi, nt, lane), c);
    }
  }
#pragma unroll
  for (int t = 0; t < TPW; ++t) {
    int nb = (nt0 + t) * 16 + ln16;
#pragma unroll
    for (int v = 0; v < 8; ++v)
      sA[(v + hi8) * APITCH + nb] = to_bf16(y[t][v]);
  }

  // ---- time loop: at top of iter k, sA holds z_k (bf16), y[] holds z_k (f32) ----
  for (int k = 0; k < Tdim; ++k) {
    __syncthreads();   // z_k visible to all waves

#if HAVE_ASYNC
    // overlap: fetch dW[k+1] while this step's GEMMs run (ping-pong buffer was
    // last read for dW[k-1], fully consumed before the barrier above)
    if (k < Tdim - 2) {
      const float* gb = dW + ((size_t)(k + 1) * Bdim + bbase) * Hdim;
      float* lb = &sDW[(k + 1) & 1][0];
#pragma unroll
      for (int i = 0; i < 8; ++i) {
        int e = tid + i * 256;
        int r = e >> 7, c4 = (e & 127) << 2;
        async_b128(gb + (size_t)r * Hdim + c4, lb + r * DWPITCH + c4);
      }
    }
#endif

    if (k < Tdim - 1) {
      for (int i = tid; i < 16 * Cdim; i += 256) {
        int r = i >> 5, c = i & 31;
        sA[r * APITCH + 512 + c] =
            to_bf16(coeffs[(size_t)(bbase + r) * Tdim * Cdim + (size_t)k * Cdim + c]);
      }
    }

    if (k >= Tdim - OUTT) {
      // ---- fused head: R = relu(z_k @ Wh1 + bh1) ----
      v8f r[TPW];
#pragma unroll
      for (int t = 0; t < TPW; ++t) {
        float bv = bh1[(nt0 + t) * 16 + ln16];
#pragma unroll
        for (int i = 0; i < 8; ++i) r[t][i] = bv;
      }
      for (int c = 0; c < 16; ++c) {
        v16bf a = loadA(sA, APITCH, lane, c * 32);
#pragma unroll
        for (int t = 0; t < TPW; ++t)
          r[t] = wmma_bf16(a, loadB(P_Wh1, c * NTILES + nt0 + t, lane), r[t]);
      }
#pragma unroll
      for (int t = 0; t < TPW; ++t) {
        int nb = (nt0 + t) * 16 + ln16;
#pragma unroll
        for (int v = 0; v < 8; ++v)
          sH[(v + hi8) * HPITCH + nb] = to_bf16(fmaxf(r[t][v], 0.0f));
      }
      __syncthreads();
      if (wave < 2) {                       // out = R @ Wh2 + bh2 (two N-tiles)
        int nt = wave;
        v8f o; float bv = bh2[nt * 16 + ln16];
#pragma unroll
        for (int i = 0; i < 8; ++i) o[i] = bv;
        for (int c = 0; c < 16; ++c) {
          v16bf a = loadA(sH, HPITCH, lane, c * 32);
          o = wmma_bf16(a, loadB(P_Wh2, c * 2 + nt, lane), o);
        }
        int tix = k - (Tdim - OUTT);
#pragma unroll
        for (int v = 0; v < 8; ++v) {
          int m = v + hi8;
          out[(size_t)(bbase + m) * OUTT * Odim + (size_t)tix * Odim + nt * 16 + ln16]
              = o[v];
        }
      }
      __syncthreads();   // sH free again; x_k visible
    } else {
      __syncthreads();   // x_k visible
    }

    if (k < Tdim - 1) {
      // ---- h = tanh([y|x] @ W1 + b1), K = 544 (17 chunks) ----
      v8f ha[TPW];
#pragma unroll
      for (int t = 0; t < TPW; ++t) {
        float bv = b1[(nt0 + t) * 16 + ln16];
#pragma unroll
        for (int i = 0; i < 8; ++i) ha[t][i] = bv;
      }
      for (int c = 0; c < 17; ++c) {
        v16bf a = loadA(sA, APITCH, lane, c * 32);
#pragma unroll
        for (int t = 0; t < TPW; ++t)
          ha[t] = wmma_bf16(a, loadB(P_W1, c * NTILES + nt0 + t, lane), ha[t]);
      }
#pragma unroll
      for (int t = 0; t < TPW; ++t) {
        int nb = (nt0 + t) * 16 + ln16;
#pragma unroll
        for (int v = 0; v < 8; ++v)
          sH[(v + hi8) * HPITCH + nb] = to_bf16(fast_tanh(ha[t][v]));
      }
      __syncthreads();

#if !HAVE_ASYNC
      // fallback path: prefetch next step's dW block (global_prefetch_b8)
      if (k < Tdim - 2) {
        const float* pf = dW + ((size_t)(k + 1) * Bdim + bbase + (tid & 15)) * Hdim
                             + (tid >> 4) * 32;
        __builtin_prefetch(pf, 0, 0);
      }
#endif

      // ---- f = tanh(h @ W2 + b2), g = sigmoid(y @ Wg + bg) ----
      v8f fa[TPW], ga[TPW];
#pragma unroll
      for (int t = 0; t < TPW; ++t) {
        float bf2 = b2[(nt0 + t) * 16 + ln16];
        float bgg = bg[(nt0 + t) * 16 + ln16];
#pragma unroll
        for (int i = 0; i < 8; ++i) { fa[t][i] = bf2; ga[t][i] = bgg; }
      }
      for (int c = 0; c < 16; ++c) {
        v16bf ah = loadA(sH, HPITCH, lane, c * 32);
        v16bf ay = loadA(sA, APITCH, lane, c * 32);
#pragma unroll
        for (int t = 0; t < TPW; ++t) {
          fa[t] = wmma_bf16(ah, loadB(P_W2, c * NTILES + nt0 + t, lane), fa[t]);
          ga[t] = wmma_bf16(ay, loadB(P_Wg, c * NTILES + nt0 + t, lane), ga[t]);
        }
      }

#if HAVE_ASYNC
      // dW[k] loads (8/thread) complete in-order before dW[k+1]'s 8: cnt<=8 is
      // enough when k+1 was issued; on the last step (k = Tdim-2, nothing issued) require cnt==0.
      if (k < Tdim - 2) wait_async_le8(); else wait_async_le0();
#endif
      __syncthreads();   // all reads of sA(z_k) done; sDW[k&1] complete + visible

      // ---- y += f*dt + g*sqdt*dW[k]; restage bf16 z_{k+1} ----
#pragma unroll
      for (int t = 0; t < TPW; ++t) {
        int nb = (nt0 + t) * 16 + ln16;
#if !HAVE_ASYNC
        const float* dwp = dW + ((size_t)k * Bdim + bbase) * Hdim + nb;
#endif
#pragma unroll
        for (int v = 0; v < 8; ++v) {
          float f = fast_tanh(fa[t][v]);
          float g = fast_sigmoid(ga[t][v]);
#if HAVE_ASYNC
          float w = sDW[k & 1][(v + hi8) * DWPITCH + nb];
#else
          float w = dwp[(size_t)(v + hi8) * Hdim];
#endif
          y[t][v] += f * dt + g * sqdt * w;
          sA[(v + hi8) * APITCH + nb] = to_bf16(y[t][v]);
        }
      }
    }
  }
}

extern "C" void kernel_launch(void* const* d_in, const int* in_sizes, int n_in,
                              void* d_out, int out_size, void* d_ws, size_t ws_size,
                              hipStream_t stream) {
  const float* times  = (const float*)d_in[0];
  const float* coeffs = (const float*)d_in[1];
  /* d_in[2] = final_index: unused by the reference computation */
  const float* dW     = (const float*)d_in[3];
  const float* W_init = (const float*)d_in[4];
  const float* b_init = (const float*)d_in[5];
  const float* W1     = (const float*)d_in[6];
  const float* b1     = (const float*)d_in[7];
  const float* W2     = (const float*)d_in[8];
  const float* b2     = (const float*)d_in[9];
  const float* Wg     = (const float*)d_in[10];
  const float* bg     = (const float*)d_in[11];
  const float* Wh1    = (const float*)d_in[12];
  const float* bh1    = (const float*)d_in[13];
  const float* Wh2    = (const float*)d_in[14];
  const float* bh2    = (const float*)d_in[15];
  (void)in_sizes; (void)n_in; (void)out_size; (void)ws_size;

  float*  hdr = (float*)d_ws;                         // [dt, sqdt]
  __bf16* pw  = (__bf16*)((char*)d_ws + 256);         // packed bf16 weights (~2.2 MB)

  sde_pack<<<512, 256, 0, stream>>>(times, W_init, W1, W2, Wg, Wh1, Wh2, hdr, pw);
  sde_main<<<Bdim / 16, 256, 0, stream>>>(coeffs, dW, b_init, b1, b2, bg, bh1, bh2,
                                          hdr, pw, (float*)d_out);
}
